// MHAlayer_72232759984137
// MI455X (gfx1250) — matrix operand order repaired and checked by reading, not compiled, not run on hardware
//
#include <hip/hip_runtime.h>

// ---------------------------------------------------------------------------
// MI455X / gfx1250 MHA layer: bf16 WMMA GEMMs with async global->LDS double
// buffering, plus LDS-resident attention. Wave32, v_wmma_f32_16x16x32_bf16.
// ---------------------------------------------------------------------------

typedef __attribute__((ext_vector_type(16))) __bf16 v16bf;
typedef __attribute__((ext_vector_type(8)))  float  v8f;

#define BATCH 2
#define SEQL  2048
#define DMODEL 1024
#define NHEAD 16
#define DK    64
#define ROWS  (BATCH * SEQL)   // 4096

__device__ __forceinline__ unsigned short f2bf(float f) {
  unsigned int u = __float_as_uint(f);
  unsigned int r = u + 0x7FFFu + ((u >> 16) & 1u);   // round-to-nearest-even
  return (unsigned short)(r >> 16);
}

// --- CDNA5 async global->LDS DMA (ASYNCcnt-tracked), via inline asm ---------
// VFLAT async form: VDST = per-lane LDS byte address, VADDR = 64-bit global
// address. Truncating a flat LDS pointer to 32 bits yields the LDS offset
// (ISA 10.2 aperture rules).
__device__ __forceinline__ void async_b128(const unsigned short* lds_dst,
                                           const unsigned short* gsrc) {
  unsigned loff = (unsigned)(unsigned long long)(const void*)lds_dst;
  unsigned long long ga = (unsigned long long)(const void*)gsrc;
  asm volatile("global_load_async_to_lds_b128 %0, %1, off"
               :: "v"(loff), "v"(ga) : "memory");
}
__device__ __forceinline__ void wait_async0() {
  asm volatile("s_wait_asynccnt 0x0" ::: "memory");
}

// A-matrix fragment (16x32 bf16 tile, row-major, stride ld elements).
// A layout (contiguous, matches B): lane m=lane&15; lanes16-31 take K+16;
// VGPR i: K=2i,2i+1.
__device__ __forceinline__ v16bf load_a_frag(const unsigned short* base, int ld, int lane) {
  int m = lane & 15, hh = (lane >> 4) & 1;
  const unsigned short* row = base + m * ld + hh * 16;
  union { v16bf v; unsigned int u[8]; } a;
#pragma unroll
  for (int i = 0; i < 8; ++i) {
    int k = i << 1;
    a.u[i] = *(const unsigned int*)(row + k);
  }
  return a.v;
}

// B-matrix fragment (K=32 x N=16), stored as 16 rows of 32 K-elements
// (i.e. row n holds B[k][n] for k=0..31). B layout (contiguous, matches A): lane n=lane&15,
// lanes 16-31 hold K=16..31, VGPR j holds K=2j,2j+1.
__device__ __forceinline__ v16bf load_b_frag(const unsigned short* base, int ld, int lane) {
  int n = lane & 15, hh = (lane >> 4) & 1;
  const unsigned short* row = base + n * ld + hh * 16;
  union { v16bf v; uint4 q[2]; } b;
  b.q[0] = *(const uint4*)(row);
  b.q[1] = *(const uint4*)(row + 8);
  return b.v;
}

// ---------------------------------------------------------------------------
// f32 -> bf16 conversion
// ---------------------------------------------------------------------------
__global__ __launch_bounds__(256) void cvt_f32_bf16(const float* __restrict__ in,
                                                    unsigned short* __restrict__ out, int n) {
  int i = blockIdx.x * blockDim.x + threadIdx.x;
  int stride = gridDim.x * blockDim.x;
  for (; i < n; i += stride) out[i] = f2bf(in[i]);
}

// ---------------------------------------------------------------------------
// C[M,N] = A[M,K] @ W[N,K]^T + bias ; bf16 inputs, f32 accumulate.
// Block tile 64x128 (8 waves, each wave 32x32 = 2x2 WMMA frags), BK=32.
// Double-buffered LDS; tiles DMA'd with global_load_async_to_lds_b128 so the
// k+1 copy overlaps the WMMAs on tile k.
// ---------------------------------------------------------------------------
template<bool STORE_F32, bool STORE_BF16>
__global__ __launch_bounds__(256)
void gemm_xwT_bf16(const unsigned short* __restrict__ A,
                   const unsigned short* __restrict__ W,
                   const float* __restrict__ bias,
                   float* __restrict__ Cf,
                   unsigned short* __restrict__ Cb,
                   int M, int N, int K) {
  constexpr int BM = 64, BN = 128, BK = 32, LDT = 40;  // 40*2B = 80B row (16B aligned)
  __shared__ __align__(16) unsigned short As[2][BM * LDT];
  __shared__ __align__(16) unsigned short Ws[2][BN * LDT];

  const int tid  = threadIdx.x;
  const int lane = tid & 31;
  const int wave = tid >> 5;
  const int wm = wave >> 2;          // 0..1
  const int wn = wave & 3;           // 0..3
  const int m0 = blockIdx.y * BM;
  const int n0 = blockIdx.x * BN;

  v8f c00 = {}, c01 = {}, c10 = {}, c11 = {};

  const int arow = tid >> 2, acol = (tid & 3) << 3;   // 64 rows x 4 chunks(8)
  const int wrow = tid >> 1, wcol = (tid & 1) << 4;   // 128 rows x 2 chunks(16)

  auto issue_tile = [&](int k0, int buf) {
    async_b128(&As[buf][arow * LDT + acol],
               &A[(size_t)(m0 + arow) * K + k0 + acol]);
    async_b128(&Ws[buf][wrow * LDT + wcol],
               &W[(size_t)(n0 + wrow) * K + k0 + wcol]);
    async_b128(&Ws[buf][wrow * LDT + wcol + 8],
               &W[(size_t)(n0 + wrow) * K + k0 + wcol + 8]);
  };

  const int nt = K / BK;
  issue_tile(0, 0);

  for (int kt = 0; kt < nt; ++kt) {
    wait_async0();          // tile kt landed in LDS (this wave's DMAs)
    __syncthreads();        // ...and everyone else's; also fences buffer reuse
    if (kt + 1 < nt) issue_tile((kt + 1) * BK, (kt + 1) & 1);

    const unsigned short* as = As[kt & 1];
    const unsigned short* ws = Ws[kt & 1];
    v16bf a0 = load_a_frag(&as[(wm * 32) * LDT],      LDT, lane);
    v16bf a1 = load_a_frag(&as[(wm * 32 + 16) * LDT], LDT, lane);
    v16bf b0 = load_b_frag(&ws[(wn * 32) * LDT],      LDT, lane);
    v16bf b1 = load_b_frag(&ws[(wn * 32 + 16) * LDT], LDT, lane);

    c00 = __builtin_amdgcn_wmma_f32_16x16x32_bf16(false, a0, false, b0, (short)0, c00, false, false);
    c01 = __builtin_amdgcn_wmma_f32_16x16x32_bf16(false, a0, false, b1, (short)0, c01, false, false);
    c10 = __builtin_amdgcn_wmma_f32_16x16x32_bf16(false, a1, false, b0, (short)0, c10, false, false);
    c11 = __builtin_amdgcn_wmma_f32_16x16x32_bf16(false, a1, false, b1, (short)0, c11, false, false);
  }

  // Epilogue. C layout: VGPR r -> M = r + 8*(lane>=16); N = lane&15.
  const int col = lane & 15;
  const int hh  = (lane >> 4) & 1;
  const int gc0 = n0 + wn * 32 + col;
  const int gc1 = gc0 + 16;
  const float bv0 = bias[gc0];
  const float bv1 = bias[gc1];
#pragma unroll
  for (int r = 0; r < 8; ++r) {
    const int rl = hh * 8 + r;
    const size_t g0 = (size_t)(m0 + wm * 32 + rl) * N;
    const size_t g1 = (size_t)(m0 + wm * 32 + 16 + rl) * N;
    float v00 = c00[r] + bv0, v01 = c01[r] + bv1;
    float v10 = c10[r] + bv0, v11 = c11[r] + bv1;
    if (STORE_F32) {
      Cf[g0 + gc0] = v00; Cf[g0 + gc1] = v01;
      Cf[g1 + gc0] = v10; Cf[g1 + gc1] = v11;
    }
    if (STORE_BF16) {
      Cb[g0 + gc0] = f2bf(v00); Cb[g0 + gc1] = f2bf(v01);
      Cb[g1 + gc0] = f2bf(v10); Cb[g1 + gc1] = f2bf(v11);
    }
  }
}

// ---------------------------------------------------------------------------
// Attention: one block per (head, batch, 16-row q tile).
// Score slab 16x2048 f32 lives in LDS (128 KB of the 320 KB WGP LDS).
// Phase 1: S = (Q Kt)/8 via WMMA.  Phase 2: softmax + write attn (f32).
// Phase 3: O = P V via WMMA (P re-quantized bf16), write bf16 context.
// ---------------------------------------------------------------------------
__global__ __launch_bounds__(256)
void attention_kernel(const unsigned short* __restrict__ qb,
                      const unsigned short* __restrict__ kb,
                      const unsigned short* __restrict__ vb,
                      float* __restrict__ attn,          // (H,B,L,L) f32
                      unsigned short* __restrict__ aob)  // (B,L,D) bf16
{
  extern __shared__ char smem[];
  float*          scores = (float*)smem;                               // [16][2048]
  unsigned short* qs     = (unsigned short*)(smem + 16 * SEQL * 4);    // [16][64]
  float*          red    = (float*)(smem + 16 * SEQL * 4 + 16 * DK * 2); // [16][16]

  const int lt = blockIdx.x, bb = blockIdx.y, hh = blockIdx.z;
  const int tid = threadIdx.x, lane = tid & 31, wave = tid >> 5;
  const int l0 = lt * 16;
  const int half = (lane >> 4) & 1, l16 = lane & 15;

  // Stage Q tile (16x64 bf16) in LDS via async DMA.
  if (tid < 128) {
    int idx = tid * 8, r = idx >> 6, c = idx & 63;
    async_b128(&qs[r * DK + c],
               &qb[((size_t)bb * SEQL + l0 + r) * DMODEL + hh * DK + c]);
  }
  wait_async0();
  __syncthreads();

  // ---- Phase 1: scores ----
  for (int ct = wave; ct < SEQL / 16; ct += 8) {
    v8f c = {};
#pragma unroll
    for (int kk = 0; kk < DK; kk += 32) {
      v16bf a = load_a_frag(qs + kk, DK, lane);
      // B = K^T tile: row n of k-tile supplies B[k][n] for k=kk..kk+31
      union { v16bf v; uint4 q[2]; } bf;
      const unsigned short* krow =
          &kb[((size_t)bb * SEQL + ct * 16 + l16) * DMODEL + hh * DK + kk + half * 16];
      bf.q[0] = *(const uint4*)krow;
      bf.q[1] = *(const uint4*)(krow + 8);
      c = __builtin_amdgcn_wmma_f32_16x16x32_bf16(false, a, false, bf.v, (short)0, c, false, false);
    }
#pragma unroll
    for (int r = 0; r < 8; ++r)
      scores[(half * 8 + r) * SEQL + ct * 16 + l16] = c[r] * 0.125f;  // 1/sqrt(64)
  }
  __syncthreads();

  // ---- Phase 2: softmax (16 threads per row) + attn output ----
  const int row = tid >> 4, sub = tid & 15;
  float* srow = scores + row * SEQL;
  float mx = -3.4e38f;
  for (int c = sub; c < SEQL; c += 16) mx = fmaxf(mx, srow[c]);
  red[row * 16 + sub] = mx;
  __syncthreads();
  mx = red[row * 16];
#pragma unroll
  for (int i = 1; i < 16; ++i) mx = fmaxf(mx, red[row * 16 + i]);
  __syncthreads();
  float sum = 0.f;
  for (int c = sub; c < SEQL; c += 16) sum += __expf(srow[c] - mx);
  red[row * 16 + sub] = sum;
  __syncthreads();
  sum = 0.f;
#pragma unroll
  for (int i = 0; i < 16; ++i) sum += red[row * 16 + i];
  const float inv = 1.0f / sum;
  float* arow = attn + (((size_t)(hh * BATCH + bb)) * SEQL + l0 + row) * SEQL;
  for (int c = sub; c < SEQL; c += 16) {
    float p = __expf(srow[c] - mx) * inv;
    srow[c] = p;        // keep normalized prob in LDS for phase 3
    arow[c] = p;        // stream attn to global (f32)
  }
  __syncthreads();

  // ---- Phase 3: O = P @ V (waves 0-3 own 16 output columns each) ----
  if (wave < 4) {
    v8f c = {};
    for (int kt = 0; kt < SEQL / 32; ++kt) {
      // A frag from LDS probs (f32 -> bf16 requantize)
      union { v16bf v; unsigned int u[8]; } a;
      const float* sr = scores + l16 * SEQL + kt * 32 + half * 16;
#pragma unroll
      for (int i = 0; i < 8; ++i) {
        int k = i << 1;
        unsigned int lo = f2bf(sr[k]);
        unsigned int hi = f2bf(sr[k + 1]);
        a.u[i] = lo | (hi << 16);
      }
      // B frag from V: element j = V[k=half*16+2j(+1)][n=l16]
      union { v16bf v; unsigned int u[8]; } bf;
      const unsigned short* vcol =
          &vb[((size_t)bb * SEQL + kt * 32 + half * 16) * DMODEL + hh * DK + wave * 16 + l16];
#pragma unroll
      for (int j = 0; j < 8; ++j) {
        unsigned int lo = vcol[(size_t)(2 * j) * DMODEL];
        unsigned int hi = vcol[(size_t)(2 * j + 1) * DMODEL];
        bf.u[j] = lo | (hi << 16);
      }
      c = __builtin_amdgcn_wmma_f32_16x16x32_bf16(false, a.v, false, bf.v, (short)0, c, false, false);
    }
#pragma unroll
    for (int r = 0; r < 8; ++r) {
      int rl = half * 8 + r;
      aob[((size_t)bb * SEQL + l0 + rl) * DMODEL + hh * DK + wave * 16 + l16] = f2bf(c[r]);
    }
  }
}

// ---------------------------------------------------------------------------
// out = LayerNorm(fcout + x) * gamma + beta   (one block per row of 1024)
// ---------------------------------------------------------------------------
__global__ __launch_bounds__(256)
void residual_ln(const float* __restrict__ fcout, const float* __restrict__ x,
                 const float* __restrict__ gamma, const float* __restrict__ beta,
                 float* __restrict__ out) {
  __shared__ float sbuf[256];
  const size_t rbase = (size_t)blockIdx.x * DMODEL;
  float y[4];
  float s = 0.f;
#pragma unroll
  for (int i = 0; i < 4; ++i) {
    int c = threadIdx.x + i * 256;
    y[i] = fcout[rbase + c] + x[rbase + c];
    s += y[i];
  }
  sbuf[threadIdx.x] = s; __syncthreads();
  for (int st = 128; st > 0; st >>= 1) {
    if (threadIdx.x < st) sbuf[threadIdx.x] += sbuf[threadIdx.x + st];
    __syncthreads();
  }
  const float mu = sbuf[0] * (1.0f / DMODEL);
  __syncthreads();
  float vs = 0.f;
#pragma unroll
  for (int i = 0; i < 4; ++i) { float d = y[i] - mu; vs += d * d; }
  sbuf[threadIdx.x] = vs; __syncthreads();
  for (int st = 128; st > 0; st >>= 1) {
    if (threadIdx.x < st) sbuf[threadIdx.x] += sbuf[threadIdx.x + st];
    __syncthreads();
  }
  const float rstd = rsqrtf(sbuf[0] * (1.0f / DMODEL) + 1e-5f);
#pragma unroll
  for (int i = 0; i < 4; ++i) {
    int c = threadIdx.x + i * 256;
    out[rbase + c] = (y[i] - mu) * rstd * gamma[c] + beta[c];
  }
}

// ---------------------------------------------------------------------------
extern "C" void kernel_launch(void* const* d_in, const int* in_sizes, int n_in,
                              void* d_out, int out_size, void* d_ws, size_t ws_size,
                              hipStream_t stream) {
  const float* x    = (const float*)d_in[0];
  const float* Wq   = (const float*)d_in[1];
  const float* bq   = (const float*)d_in[2];
  const float* Wk   = (const float*)d_in[3];
  const float* bk   = (const float*)d_in[4];
  const float* Wv   = (const float*)d_in[5];
  const float* bv   = (const float*)d_in[6];
  const float* Wfc  = (const float*)d_in[7];
  const float* bfc  = (const float*)d_in[8];
  const float* gamma= (const float*)d_in[9];
  const float* beta = (const float*)d_in[10];
  (void)in_sizes; (void)n_in; (void)out_size; (void)ws_size;

  const size_t NX = (size_t)ROWS * DMODEL;     // 4,194,304
  const size_t NW = (size_t)DMODEL * DMODEL;   // 1,048,576

  char* p = (char*)d_ws;
  auto alloc = [&](size_t bytes) { char* r = p; p += (bytes + 255) & ~(size_t)255; return r; };
  unsigned short* xb   = (unsigned short*)alloc(NX * 2);
  unsigned short* wqb  = (unsigned short*)alloc(NW * 2);
  unsigned short* wkb  = (unsigned short*)alloc(NW * 2);
  unsigned short* wvb  = (unsigned short*)alloc(NW * 2);
  unsigned short* wfcb = (unsigned short*)alloc(NW * 2);
  unsigned short* t1b  = (unsigned short*)alloc(NX * 2);
  unsigned short* q_b  = (unsigned short*)alloc(NX * 2);
  unsigned short* k_b  = (unsigned short*)alloc(NX * 2);
  unsigned short* v_b  = (unsigned short*)alloc(NX * 2);
  unsigned short* aob  = (unsigned short*)alloc(NX * 2);
  float*          fco  = (float*)alloc(NX * 4);

  // 1) bf16 conversions
  cvt_f32_bf16<<<4096, 256, 0, stream>>>(x,   xb,   (int)NX);
  cvt_f32_bf16<<<1024, 256, 0, stream>>>(Wq,  wqb,  (int)NW);
  cvt_f32_bf16<<<1024, 256, 0, stream>>>(Wk,  wkb,  (int)NW);
  cvt_f32_bf16<<<1024, 256, 0, stream>>>(Wv,  wvb,  (int)NW);
  cvt_f32_bf16<<<1024, 256, 0, stream>>>(Wfc, wfcb, (int)NW);

  // 2) double projections (proj2): t = (x W^T + b) W^T + b
  dim3 gg(DMODEL / 128, ROWS / 64);  // (8, 64)
  gemm_xwT_bf16<false, true><<<gg, 256, 0, stream>>>(xb,  wqb, bq, nullptr, t1b, ROWS, DMODEL, DMODEL);
  gemm_xwT_bf16<false, true><<<gg, 256, 0, stream>>>(t1b, wqb, bq, nullptr, q_b, ROWS, DMODEL, DMODEL);
  gemm_xwT_bf16<false, true><<<gg, 256, 0, stream>>>(xb,  wkb, bk, nullptr, t1b, ROWS, DMODEL, DMODEL);
  gemm_xwT_bf16<false, true><<<gg, 256, 0, stream>>>(t1b, wkb, bk, nullptr, k_b, ROWS, DMODEL, DMODEL);
  gemm_xwT_bf16<false, true><<<gg, 256, 0, stream>>>(xb,  wvb, bv, nullptr, t1b, ROWS, DMODEL, DMODEL);
  gemm_xwT_bf16<false, true><<<gg, 256, 0, stream>>>(t1b, wvb, bv, nullptr, v_b, ROWS, DMODEL, DMODEL);

  // 3) attention (scores + softmax + attn write + P@V)
  float* out  = (float*)d_out;
  float* attn = out + NX;  // tuple: (out, attn) concatenated
  size_t smem = 16 * SEQL * 4 + 16 * DK * 2 + 16 * 16 * 4;  // ~131 KB < 320 KB WGP LDS
  attention_kernel<<<dim3(SEQL / 16, BATCH, NHEAD), 256, smem, stream>>>(q_b, k_b, v_b, attn, aob);

  // 4) FC projection (f32 out) + residual + LayerNorm
  gemm_xwT_bf16<true, false><<<gg, 256, 0, stream>>>(aob, wfcb, bfc, fco, nullptr, ROWS, DMODEL, DMODEL);
  residual_ln<<<ROWS, 256, 0, stream>>>(fco, x, gamma, beta, out);
}
